// TFPhi3Attention_89378269430242
// MI455X (gfx1250) — matrix operand order repaired and checked by reading, not compiled, not run on hardware
//
#include <hip/hip_runtime.h>
#include <hip/hip_bf16.h>

#ifndef __has_builtin
#define __has_builtin(x) 0
#endif

// ---------------- CDNA5 WMMA / TDM types ----------------
typedef __attribute__((ext_vector_type(16))) __bf16 v16bf;
typedef __attribute__((ext_vector_type(8)))  __bf16 v8bf;
typedef __attribute__((ext_vector_type(8)))  float  v8f;
typedef __attribute__((ext_vector_type(4)))  unsigned int u32x4;
typedef __attribute__((ext_vector_type(8)))  int i32x8;
typedef __attribute__((ext_vector_type(4)))  int i32x4;

#if __has_builtin(__builtin_amdgcn_tensor_load_to_lds) && __has_builtin(__builtin_amdgcn_s_wait_tensorcnt)
#define HAVE_TDM 1
#pragma message("CDNA5 probe: TDM tensor_load_to_lds path ENABLED")
#else
#define HAVE_TDM 0
#pragma message("CDNA5 probe: TDM path DISABLED (builtin not declared)")
#endif

union FragBF { v16bf v; v8bf h[2]; };

__device__ __forceinline__ v8f wmma_bf16(v16bf a, v16bf b, v8f c) {
  // D = A(16x32 bf16) * B(32x16 bf16) + C(16x16 f32)
  return __builtin_amdgcn_wmma_f32_16x16x32_bf16(
      false, a, false, b, (short)0, c, false, false);
}

// fp32 -> bf16 (round to nearest even)
__device__ __forceinline__ unsigned short f2bf(float f) {
  union { float f; unsigned u; } x; x.f = f;
  unsigned r = x.u + 0x7FFFu + ((x.u >> 16) & 1u);
  return (unsigned short)(r >> 16);
}

// A fragment: 16x32 bf16 tile at (row0, k0) of row-major [*, stride] bf16.
__device__ __forceinline__ v16bf load_frag_a(const unsigned short* base, int stride,
                                             int row0, int k0, int lane) {
  int r = lane & 15, hf = lane >> 4;
  const unsigned short* p = base + (size_t)(row0 + r) * stride + k0 + hf * 8;
  FragBF f;
  f.h[0] = *(const v8bf*)(p);
  f.h[1] = *(const v8bf*)(p + 16);
  return f.v;
}

// B fragment: 32x16 tile, cols n0..n0+15, K k0..k0+31, matrix stored [N][K].
__device__ __forceinline__ v16bf load_frag_b(const unsigned short* baseT, int ldk,
                                             int n0, int k0, int lane) {
  int r = lane & 15, hf = lane >> 4;
  const unsigned short* p = baseT + (size_t)(n0 + r) * ldk + k0 + hf * 8;
  FragBF f;
  f.h[0] = *(const v8bf*)(p);
  f.h[1] = *(const v8bf*)(p + 16);
  return f.v;
}

// ---------------- problem constants ----------------
#define Bq 2
#define Sq 2048
#define HIDq 3072
#define NH 32
#define NKV 8
#define HD 96
#define OPSZ 4608
#define MROWS 4096
#define LOG_THETA 9.210340371976184f
#define ATTN_SCALE 0.10206207261596577f // 1/sqrt(96)

// ---------------- TDM: DMA a 64x96 bf16 tile (row-major) into LDS ----------
#if HAVE_TDM
__device__ __forceinline__ void tdm_load_k_tile(const unsigned short* gsrc, void* ldsDst) {
  unsigned lds_addr = (unsigned)(size_t)ldsDst;
  unsigned long long ga = (unsigned long long)(size_t)gsrc;
  // D# group0: [1:0]=count=1, [63:32]=lds_addr, [120:64]=global_addr, [127:126]=type 2
  u32x4 g0 = { 1u, lds_addr, (unsigned)ga,
               ((unsigned)(ga >> 32) & 0x01FFFFFFu) | 0x80000000u };
  // D# group1: data_size=1(2B) @[17:16]; tensor_dim0=96 @[79:48];
  // tensor_dim1=64 @[111:80]; tile_dim0=96 @[127:112]; tile_dim1=64 @[143:128];
  // tensor_dim0_stride=96 @[207:160]; tensor_dim1_stride=0
  i32x8 g1 = { 0x00010000, (int)(96u << 16), (int)(64u << 16), (int)(96u << 16),
               64, 96, 0, 0 };
  i32x4 g2 = { 0, 0, 0, 0 };
  i32x4 g3 = { 0, 0, 0, 0 };
#if defined(__clang_major__) && (__clang_major__ >= 23)
  i32x8 g4 = { 0, 0, 0, 0, 0, 0, 0, 0 };
  __builtin_amdgcn_tensor_load_to_lds(g0, g1, g2, g3, g4, 0);
#else
  __builtin_amdgcn_tensor_load_to_lds(g0, g1, g2, g3, 0);
#endif
}
#endif

// ---------------- conversion kernels ----------------
__global__ void k_conv_bf16(const float* __restrict__ in,
                            unsigned short* __restrict__ out, size_t n) {
  size_t i = (size_t)blockIdx.x * 256 + threadIdx.x;
  if (i < n) out[i] = f2bf(in[i]);
}

// in: [R][C] f32  ->  out: [C][R] bf16
__global__ void k_transpose_bf16(const float* __restrict__ in,
                                 unsigned short* __restrict__ out, int R, int C) {
  size_t i = (size_t)blockIdx.x * 256 + threadIdx.x;
  size_t total = (size_t)R * C;
  if (i < total) {
    int c = (int)(i / R);
    int r = (int)(i % R);
    out[i] = f2bf(in[(size_t)r * C + c]);
  }
}

// ---------------- QKV GEMM + RoPE ----------------
// Each wave: 32(M) x 96(N) strip -> B fragments reused twice, RoPE pairing
// d <-> d+48 stays in-register (tile t <-> t+3, same lane/VGPR).
__global__ void __launch_bounds__(256, 1)
k_qkv_rope(const unsigned short* __restrict__ X,      // [4096][3072]
           const unsigned short* __restrict__ WqkvT,  // [4608][3072]
           unsigned short* __restrict__ Qb,           // [B][32][S][96]
           unsigned short* __restrict__ Kb,           // [B][8][S][96]
           unsigned short* __restrict__ Vb) {         // [B][8][S][96]
  int lane = threadIdx.x & 31;
  int wave = threadIdx.x >> 5;
  int task = blockIdx.x * 8 + wave;      // 6144 tasks = 48 strips * 128 m-tiles
  int nstrip = task >> 7;                // 0..47
  int m0 = (task & 127) << 5;            // 32-row tiles
  int c0 = nstrip * HD;

  const unsigned short* pfA = X + (size_t)(m0 + lane) * HIDq;

  v8f acc[2][6] = {};
  for (int k0 = 0; k0 < HIDq; k0 += 32) {
    if (k0 + 64 < HIDq) __builtin_prefetch(pfA + k0 + 64, 0, 3);
    v16bf a0 = load_frag_a(X, HIDq, m0, k0, lane);
    v16bf a1 = load_frag_a(X, HIDq, m0 + 16, k0, lane);
#pragma unroll
    for (int t = 0; t < 6; ++t) {
      v16bf bf = load_frag_b(WqkvT, HIDq, c0 + t * 16, k0, lane);
      acc[0][t] = wmma_bf16(a0, bf, acc[0][t]);
      acc[1][t] = wmma_bf16(a1, bf, acc[1][t]);
    }
  }

  int hf = lane >> 4, r = lane & 15;

  if (nstrip < 40) {  // RoPE on Q and K strips
#pragma unroll
    for (int t = 0; t < 3; ++t) {
      float fidx = (float)(t * 16 + r);
      float inv = __expf(-(2.0f * fidx / (float)HD) * LOG_THETA);
#pragma unroll
      for (int mi = 0; mi < 2; ++mi)
#pragma unroll
        for (int v = 0; v < 8; ++v) {
          int m = m0 + mi * 16 + v + hf * 8;
          int s = m & (Sq - 1);
          float ang = (float)s * inv;
          float sn, cs;
          __sincosf(ang, &sn, &cs);
          float x1 = acc[mi][t][v], x2 = acc[mi][t + 3][v];
          acc[mi][t][v]     = x1 * cs - x2 * sn;
          acc[mi][t + 3][v] = x2 * cs + x1 * sn;
        }
    }
  }

#pragma unroll
  for (int t = 0; t < 6; ++t) {
    int c = c0 + t * 16 + r;
#pragma unroll
    for (int mi = 0; mi < 2; ++mi)
#pragma unroll
      for (int v = 0; v < 8; ++v) {
        int m = m0 + mi * 16 + v + hf * 8;
        int b = m >> 11, s = m & (Sq - 1);
        unsigned short val = f2bf(acc[mi][t][v]);
        if (c < NH * HD) {
          int h = c / HD, d = c % HD;
          Qb[(((size_t)b * NH + h) * Sq + s) * HD + d] = val;
        } else if (c < NH * HD + NKV * HD) {
          int cc = c - NH * HD;
          int h = cc / HD, d = cc % HD;
          Kb[(((size_t)b * NKV + h) * Sq + s) * HD + d] = val;
        } else {
          int cc = c - NH * HD - NKV * HD;
          int h = cc / HD, d = cc % HD;
          Vb[(((size_t)b * NKV + h) * Sq + s) * HD + d] = val;
        }
      }
  }
}

// ---------------- flash attention ----------------
__global__ void __launch_bounds__(256, 1)
k_flash_attn(const unsigned short* __restrict__ Qb,
             const unsigned short* __restrict__ Kb,
             const unsigned short* __restrict__ Vb,
             unsigned short* __restrict__ AO) { // [B][S][3072] bf16
  __shared__ __align__(16) unsigned short Ks[64 * HD];    // [key][d]
  __shared__ __align__(16) unsigned short Vt[HD * 64];    // [d][key]
  __shared__ __align__(16) unsigned short Pb[8][16 * 64]; // per-wave P scratch

  int lane = threadIdx.x & 31;
  int wave = threadIdx.x >> 5;
  int hf = lane >> 4, r = lane & 15;

  int mb = blockIdx.x & 15;
  int h  = (blockIdx.x >> 4) & 31;
  int b  = blockIdx.x >> 9;
  int hk = h >> 2;  // GQA

  const unsigned short* Qg = Qb + ((size_t)b * NH + h) * Sq * HD;
  const unsigned short* Kg = Kb + ((size_t)b * NKV + hk) * Sq * HD;
  const unsigned short* Vg = Vb + ((size_t)b * NKV + hk) * Sq * HD;

  int qbase = mb * 128;
  int qw = qbase + wave * 16;

  v16bf qf[3];
#pragma unroll
  for (int kk = 0; kk < 3; ++kk) qf[kk] = load_frag_a(Qg, HD, qw, kk * 32, lane);

  v8f o[6] = {};
  float mrun[8], lrun[8];
#pragma unroll
  for (int v = 0; v < 8; ++v) { mrun[v] = -1e30f; lrun[v] = 0.0f; }

  int kend = qbase + 128;  // causal
  for (int kb = 0; kb < kend; kb += 64) {
    __syncthreads();
#if HAVE_TDM
    if (wave == 0) tdm_load_k_tile(Kg + (size_t)kb * HD, (void*)Ks);
    for (int i = threadIdx.x; i < 64 * HD; i += 256) {
      int kr = i / HD, kc = i % HD;
      Vt[kc * 64 + kr] = Vg[(size_t)(kb + kr) * HD + kc];
    }
    if (wave == 0) __builtin_amdgcn_s_wait_tensorcnt(0);
#else
    for (int i = threadIdx.x; i < 64 * HD; i += 256) {
      int kr = i / HD, kc = i % HD;
      unsigned short kvK = Kg[(size_t)(kb + kr) * HD + kc];
      unsigned short kvV = Vg[(size_t)(kb + kr) * HD + kc];
      Ks[i] = kvK;
      Vt[kc * 64 + kr] = kvV;
    }
#endif
    __syncthreads();

    // scores 16x64 = Q(16x96) @ K^T(96x64)
    v8f sc[4] = {};
#pragma unroll
    for (int t = 0; t < 4; ++t)
#pragma unroll
      for (int kk = 0; kk < 3; ++kk)
        sc[t] = wmma_bf16(qf[kk], load_frag_b(Ks, HD, t * 16, kk * 32, lane), sc[t]);

    // online softmax
    unsigned short* Pw = Pb[wave];
#pragma unroll
    for (int v = 0; v < 8; ++v) {
      int q = qw + v + hf * 8;
      float mx = -1e30f;
#pragma unroll
      for (int t = 0; t < 4; ++t) {
        int kn = kb + t * 16 + r;
        float x = sc[t][v] * ATTN_SCALE;
        x = (kn <= q) ? x : -1e30f;
        sc[t][v] = x;
        mx = fmaxf(mx, x);
      }
#pragma unroll
      for (int off = 1; off < 16; off <<= 1)
        mx = fmaxf(mx, __shfl_xor(mx, off, 32));
      float mnew = fmaxf(mrun[v], mx);
      float alpha = __expf(mrun[v] - mnew);
      float rs = 0.0f;
#pragma unroll
      for (int t = 0; t < 4; ++t) {
        float p = __expf(sc[t][v] - mnew);
        sc[t][v] = p;
        rs += p;
      }
#pragma unroll
      for (int off = 1; off < 16; off <<= 1)
        rs += __shfl_xor(rs, off, 32);
      lrun[v] = lrun[v] * alpha + rs;
      mrun[v] = mnew;
#pragma unroll
      for (int u = 0; u < 6; ++u) o[u][v] *= alpha;
    }

    // C-layout -> A-layout transpose of P via per-wave LDS scratch
#pragma unroll
    for (int t = 0; t < 4; ++t)
#pragma unroll
      for (int v = 0; v < 8; ++v)
        Pw[(v + hf * 8) * 64 + t * 16 + r] = f2bf(sc[t][v]);

    // O += P(16x64) @ V(64x96)
#pragma unroll
    for (int ks = 0; ks < 2; ++ks) {
      v16bf pa = load_frag_a(Pw, 64, 0, ks * 32, lane);
#pragma unroll
      for (int u = 0; u < 6; ++u) {
        v16bf vb = load_frag_b(Vt, 64, u * 16, ks * 32, lane);
        o[u] = wmma_bf16(pa, vb, o[u]);
      }
    }
  }

#pragma unroll
  for (int v = 0; v < 8; ++v) {
    float inv = 1.0f / lrun[v];
    int q = qw + v + hf * 8;
#pragma unroll
    for (int u = 0; u < 6; ++u) {
      int d = u * 16 + r;
      AO[((size_t)b * Sq + q) * HIDq + h * HD + d] = f2bf(o[u][v] * inv);
    }
  }
}

// ---------------- output GEMM: out = AO @ Wo (f32 out) ----------------
__global__ void __launch_bounds__(256, 1)
k_out_gemm(const unsigned short* __restrict__ A,    // [4096][3072]
           const unsigned short* __restrict__ WoT,  // [3072][3072]
           float* __restrict__ out) {               // [4096][3072] f32
  int lane = threadIdx.x & 31;
  int wave = threadIdx.x >> 5;
  int task = blockIdx.x * 8 + wave;  // 4096 tasks = 32 strips * 128 m-tiles
  int nstrip = task >> 7;
  int m0 = (task & 127) << 5;
  int c0 = nstrip * HD;

  const unsigned short* pfA = A + (size_t)(m0 + lane) * HIDq;

  v8f acc[2][6] = {};
  for (int k0 = 0; k0 < HIDq; k0 += 32) {
    if (k0 + 64 < HIDq) __builtin_prefetch(pfA + k0 + 64, 0, 3);
    v16bf a0 = load_frag_a(A, HIDq, m0, k0, lane);
    v16bf a1 = load_frag_a(A, HIDq, m0 + 16, k0, lane);
#pragma unroll
    for (int t = 0; t < 6; ++t) {
      v16bf bf = load_frag_b(WoT, HIDq, c0 + t * 16, k0, lane);
      acc[0][t] = wmma_bf16(a0, bf, acc[0][t]);
      acc[1][t] = wmma_bf16(a1, bf, acc[1][t]);
    }
  }

  int hf = lane >> 4, r = lane & 15;
#pragma unroll
  for (int t = 0; t < 6; ++t)
#pragma unroll
    for (int mi = 0; mi < 2; ++mi)
#pragma unroll
      for (int v = 0; v < 8; ++v)
        out[(size_t)(m0 + mi * 16 + v + hf * 8) * HIDq + c0 + t * 16 + r] = acc[mi][t][v];
}

// ---------------- launch ----------------
extern "C" void kernel_launch(void* const* d_in, const int* in_sizes, int n_in,
                              void* d_out, int out_size, void* d_ws, size_t ws_size,
                              hipStream_t stream) {
  const float* hidden = (const float*)d_in[0];
  // d_in[1] position_ids: arange(S), recomputed on device
  // d_in[2] attention_mask: causal, applied analytically
  const float* Wqkv = (const float*)d_in[3];
  const float* Wo   = (const float*)d_in[4];
  float* out = (float*)d_out;

  char* ws = (char*)d_ws;
  size_t off = 0;
  auto alloc = [&](size_t bytes) {
    void* p = ws + off;
    off += (bytes + 255) & ~(size_t)255;
    return p;
  };
  unsigned short* Xb    = (unsigned short*)alloc((size_t)MROWS * HIDq * 2);
  unsigned short* WqkvT = (unsigned short*)alloc((size_t)OPSZ * HIDq * 2);
  unsigned short* WoT   = (unsigned short*)alloc((size_t)HIDq * HIDq * 2);
  unsigned short* Qb    = (unsigned short*)alloc((size_t)Bq * NH * Sq * HD * 2);
  unsigned short* Kb    = (unsigned short*)alloc((size_t)Bq * NKV * Sq * HD * 2);
  unsigned short* Vb    = (unsigned short*)alloc((size_t)Bq * NKV * Sq * HD * 2);
  unsigned short* AO    = (unsigned short*)alloc((size_t)MROWS * HIDq * 2);

  size_t nX = (size_t)MROWS * HIDq;
  k_conv_bf16<<<(unsigned)((nX + 255) / 256), 256, 0, stream>>>(hidden, Xb, nX);

  size_t nW1 = (size_t)HIDq * OPSZ;
  k_transpose_bf16<<<(unsigned)((nW1 + 255) / 256), 256, 0, stream>>>(Wqkv, WqkvT, HIDq, OPSZ);

  size_t nW2 = (size_t)HIDq * HIDq;
  k_transpose_bf16<<<(unsigned)((nW2 + 255) / 256), 256, 0, stream>>>(Wo, WoT, HIDq, HIDq);

  // 48 n-strips * 128 m-tiles = 6144 wave-tasks / 8 per block
  k_qkv_rope<<<768, 256, 0, stream>>>(Xb, WqkvT, Qb, Kb, Vb);

  // B * NH * (S/128) = 1024 blocks
  k_flash_attn<<<1024, 256, 0, stream>>>(Qb, Kb, Vb, AO);

  // 32 n-strips * 128 m-tiles = 4096 wave-tasks / 8 per block
  k_out_gemm<<<512, 256, 0, stream>>>(AO, WoT, out);
}
